// TripleAdaptiveQuantizerV5_89781996356230
// MI455X (gfx1250) — matrix operand order repaired and checked by reading, not compiled, not run on hardware
//
#include <hip/hip_runtime.h>
#include <math.h>

// ---------------- problem constants ----------------
#define B_   64
#define L_   256
#define N_   8
#define K_   64
#define HD_  64
#define M_TOTAL (B_*L_*K_)          // 1,048,576 rows of the fused MLP
#define ROWS_PER_BLOCK 128          // 8 waves x 16 rows
#define CHUNKS_PER_BLOCK 8
#define NBLOCKS (M_TOTAL / (ROWS_PER_BLOCK * CHUNKS_PER_BLOCK))   // 1024

// output offsets (floats), concatenated in reference return order
#define O_VQ  0ull           // (B,L,K,2)      2,097,152
#define O_HQ  2097152ull     // (B,L,N,K,2)   16,777,216
#define O_YQ  18874368ull    // (B,L,N,2)        262,144
#define O_EB  19136512ull    // (B,L,K)        1,048,576
#define O_WY  20185088ull    // (B,L,6)           98,304
#define O_WH  20283392ull    // (B,L,K,6)      6,291,456
#define O_WD  26574848ull    // (B,L,K,6)      6,291,456
#define O_YB  32866304ull    // (B,L)             16,384
#define O_HB  32882688ull    // (B,L,K)        1,048,576
#define O_DB  33931264ull    // (B,L,K)        1,048,576

typedef __attribute__((ext_vector_type(16))) _Float16 v16h;
typedef __attribute__((ext_vector_type(8)))  _Float16 half8;
typedef __attribute__((ext_vector_type(8)))  float    v8f;

// ---------------- small helpers ----------------
__device__ __forceinline__ unsigned mix32(unsigned x) {
    x ^= x >> 16; x *= 0x7feb352dU;
    x ^= x >> 15; x *= 0x846ca68bU;
    x ^= x >> 16;
    return x;
}
// deterministic stand-in for jax threefry gumbel noise (no runtime check available)
__device__ __forceinline__ float gumbelN(unsigned stream, unsigned idx) {
    unsigned h = mix32(idx * 0x9E3779B9U + stream * 0x85EBCA6BU + 0x165667B1U);
    float u = ((float)(h >> 8) + 0.5f) * (1.0f / 16777216.0f);   // strictly (0,1)
    return -logf(-logf(u));
}
__device__ __forceinline__ float lsqf(float x, float s, float qn, float qp) {
    float t = x / s;
    t = fminf(fmaxf(t, qn), qp);
    return rintf(t) * s;   // round-half-even like jnp.round
}

// A fragment (16x32 f16, MxK): lane m=lane%16, g=lane/16.
// VGPR0..3 hold K = kbase+g*8 .. +7 ; VGPR4..7 hold K = kbase+16+g*8 .. +7
__device__ __forceinline__ v16h loadA(const _Float16* base, int stride, int m, int g, int kbase) {
    const _Float16* row = base + m * stride + kbase + g * 8;
    union { v16h v; half8 h[2]; } u;
    u.h[0] = *(const half8*)(row);
    u.h[1] = *(const half8*)(row + 16);
    return u.v;
}
// B fragment (32x16 f16, KxN), weights stored [n][k]: lane holds column n=lane%16,
// K = kbase+g*8 .. +7, then kbase+16+g*8 .. +7 (same order as A).
__device__ __forceinline__ v16h loadB(const _Float16* base, int stride, int n, int g, int kbase) {
    const _Float16* row = base + n * stride + kbase + g * 8;
    union { v16h v; half8 h[2]; } u;
    u.h[0] = *(const half8*)(row);
    u.h[1] = *(const half8*)(row + 16);
    return u.v;
}
__device__ __forceinline__ v8f wmma16(v16h a, v16h b, v8f c) {
    return __builtin_amdgcn_wmma_f32_16x16x32_f16(false, a, false, b, (short)0, c, false, false);
}

// ---------------- kernel 1: per-(b,l) power stats ----------------
__global__ __launch_bounds__(64) void taq_prep_kernel(
    const float* __restrict__ H, const float* __restrict__ y,
    float* __restrict__ ws_hp, float* __restrict__ ws_tp, float* __restrict__ ws_yp)
{
    __shared__ float sred[K_];
    const int bl = blockIdx.x;          // 0 .. B*L-1
    const int k  = threadIdx.x;         // 0 .. 63
    const float2* Hp = (const float2*)H + (size_t)bl * (N_ * K_);
    float hp = 0.f;
#pragma unroll
    for (int n = 0; n < N_; ++n) {
        float2 t = Hp[n * K_ + k];
        hp += t.x * t.x + t.y * t.y;
    }
    ws_hp[(size_t)bl * K_ + k] = hp;
    sred[k] = hp;
    __syncthreads();
    if (k == 0) {
        float tp = 0.f;
        for (int i = 0; i < K_; ++i) tp += sred[i];
        ws_tp[bl] = tp;
        const float2* yp = (const float2*)y + (size_t)bl * N_;
        float ys = 0.f;
#pragma unroll
        for (int n = 0; n < N_; ++n) { float2 t = yp[n]; ys += t.x * t.x + t.y * t.y; }
        ws_yp[bl] = log1pf(ys);
    }
}

// ---------------- kernel 2: mean of h_power over L -> (b,k) ----------------
__global__ __launch_bounds__(256) void taq_avg_kernel(
    const float* __restrict__ ws_hp, float* __restrict__ ws_avg)
{
    int t = blockIdx.x * blockDim.x + threadIdx.x;   // 0 .. 4095
    int b = t >> 6, k = t & 63;
    const float* p = ws_hp + (size_t)b * (L_ * K_) + k;
    float s = 0.f;
    for (int l = 0; l < L_; ++l) s += p[l * K_];
    ws_avg[t] = s * (1.0f / (float)L_);
}

// ---------------- kernel 3: fused MLP (WMMA) + gumbel + quantize ----------------
__global__ __launch_bounds__(256) void taq_main_kernel(
    const float* __restrict__ v,  const float* __restrict__ H,  const float* __restrict__ y,
    const float* __restrict__ snr,
    const float* __restrict__ w1, const float* __restrict__ b1,
    const float* __restrict__ w2, const float* __restrict__ b2,
    const float* __restrict__ w3, const float* __restrict__ b3,
    const float* __restrict__ wy, const float* __restrict__ by,
    const float* __restrict__ wh, const float* __restrict__ bh,
    const float* __restrict__ wd, const float* __restrict__ bd,
    const float* __restrict__ s_y, const float* __restrict__ s_H, const float* __restrict__ s_d,
    const int*   __restrict__ tau_p,
    const float* __restrict__ ws_hp, const float* __restrict__ ws_tp,
    const float* __restrict__ ws_yp, const float* __restrict__ ws_avg,
    float* __restrict__ out)
{
    // weight B-matrices stored [n][k] f16 (padded, 16B-aligned row strides for b128 LDS loads)
    __shared__ __attribute__((aligned(16))) _Float16 sB1[64][40];   // K pad 9->32
    __shared__ __attribute__((aligned(16))) _Float16 sB2[64][80];
    __shared__ __attribute__((aligned(16))) _Float16 sB3[64][80];
    __shared__ __attribute__((aligned(16))) _Float16 sBh[32][80];   // wy|wh|wd rows 0..17
    __shared__ __attribute__((aligned(16))) _Float16 sStag[8][16][80]; // per-wave A/act staging; aliased f32 logits
    __shared__ float sBias1[64], sBias2[64], sBias3[64], sBiasH[32];
    __shared__ float sScale[18];          // s_y[6] | s_H[6] | s_d[6]
    __shared__ float sBitsTab[6], sQPTab[6], sQNTab[6];
    __shared__ float sLyAcc[2][6];
    __shared__ int   sIdxH[128], sIdxD[128], sIY[2];
    __shared__ float sYbits[2];

    const int tid  = threadIdx.x;
    const int wave = tid >> 5, lane = tid & 31;
    const int lrow = lane & 15, g = lane >> 4;

    // ---- one-time weight staging ----
    for (int idx = tid; idx < 64 * 32; idx += 256) {
        int n = idx >> 5, kk = idx & 31;
        sB1[n][kk] = (kk < 9) ? (_Float16)w1[n * 9 + kk] : (_Float16)0.f;
    }
    for (int idx = tid; idx < 64 * 64; idx += 256) {
        int n = idx >> 6, kk = idx & 63;
        sB2[n][kk] = (_Float16)w2[idx];
        sB3[n][kk] = (_Float16)w3[idx];
    }
    for (int idx = tid; idx < 32 * 64; idx += 256) {
        int n = idx >> 6, kk = idx & 63;
        float val = 0.f;
        if (n < 6)       val = wy[n * 64 + kk];
        else if (n < 12) val = wh[(n - 6) * 64 + kk];
        else if (n < 18) val = wd[(n - 12) * 64 + kk];
        sBh[n][kk] = (_Float16)val;
    }
    if (tid < 64) { sBias1[tid] = b1[tid]; sBias2[tid] = b2[tid]; sBias3[tid] = b3[tid]; }
    if (tid < 32) {
        float val = 0.f;
        if (tid < 6)       val = by[tid];
        else if (tid < 12) val = bh[tid - 6];
        else if (tid < 18) val = bd[tid - 12];
        sBiasH[tid] = val;
    }
    if (tid < 18) sScale[tid] = (tid < 6) ? s_y[tid] : ((tid < 12) ? s_H[tid - 6] : s_d[tid - 12]);
    if (tid < 6) {
        const float bitsv[6] = {2.f, 4.f, 6.f, 8.f, 12.f, 16.f};
        sBitsTab[tid] = bitsv[tid];
        float qp = (float)((1 << ((int)bitsv[tid] - 1)) - 1);
        sQPTab[tid] = qp;
        sQNTab[tid] = -(qp + 1.0f);
    }
    __syncthreads();

    int ti = tau_p[0];
    const float tauInv = 1.0f / ((ti == 0) ? 1.0f : (float)ti);

    float* out_vq = out + O_VQ;  float* out_Hq = out + O_HQ;  float* out_yq = out + O_YQ;
    float* out_eb = out + O_EB;  float* out_wy = out + O_WY;  float* out_wH = out + O_WH;
    float* out_wd = out + O_WD;  float* out_yb = out + O_YB;  float* out_Hb = out + O_HB;
    float* out_db = out + O_DB;

    for (int chunk = 0; chunk < CHUNKS_PER_BLOCK; ++chunk) {
        const int rowBase = (blockIdx.x * CHUNKS_PER_BLOCK + chunk) * ROWS_PER_BLOCK;
        const int blBase  = rowBase >> 6;          // first (b,l) group (block covers 2)
        const int waveRow = rowBase + wave * 16;

        if (tid < 12) sLyAcc[tid / 6][tid % 6] = 0.f;
        __syncthreads();

        // ---- features: lanes 0..15 each build one row (9 feats, zero pad to 32) ----
        if (lane < 16) {
            int r  = waveRow + lane;
            int bl = r >> 6, k = r & 63, b = r >> 14;
            float2 vv = ((const float2*)v)[r];
            float sn  = snr[r];
            float hp  = ws_hp[r];
            float av  = ws_avg[(b << 6) + k];
            float tp  = ws_tp[bl];
            float yw  = ws_yp[bl];
            float sir = log1pf(hp / (tp - hp + 1e-10f));
            float tpn = log1pf(tp);
            float sp  = sqrtf(vv.x * vv.x + vv.y * vv.y + 1e-10f);
            _Float16* dst = &sStag[wave][lane][0];
            dst[0] = (_Float16)vv.x; dst[1] = (_Float16)vv.y; dst[2] = (_Float16)sn;
            dst[3] = (_Float16)hp;   dst[4] = (_Float16)av;   dst[5] = (_Float16)sir;
            dst[6] = (_Float16)tpn;  dst[7] = (_Float16)yw;   dst[8] = (_Float16)sp;
#pragma unroll
            for (int j = 9; j < 32; ++j) dst[j] = (_Float16)0.f;
        }

        const _Float16* stag  = &sStag[wave][0][0];
        _Float16*       stagw = &sStag[wave][0][0];

        // ---- layer 1: 9(pad32) -> 64, 4 WMMAs ----
        {
            v16h a = loadA(stag, 80, lrow, g, 0);
#pragma unroll
            for (int t = 0; t < 4; ++t) {
                v8f c = {};
                c = wmma16(a, loadB(&sB1[0][0], 40, t * 16 + lrow, g, 0), c);
                int n = t * 16 + lrow;
#pragma unroll
                for (int rr = 0; rr < 8; ++rr) {
                    float hv = fmaxf(c[rr] + sBias1[n], 0.f);
                    stagw[(rr + 8 * g) * 80 + n] = (_Float16)hv;
                }
            }
        }
        // ---- layer 2: 64 -> 64, 8 WMMAs ----
        {
            v16h alo = loadA(stag, 80, lrow, g, 0);
            v16h ahi = loadA(stag, 80, lrow, g, 32);
#pragma unroll
            for (int t = 0; t < 4; ++t) {
                v8f c = {};
                c = wmma16(alo, loadB(&sB2[0][0], 80, t * 16 + lrow, g, 0), c);
                c = wmma16(ahi, loadB(&sB2[0][0], 80, t * 16 + lrow, g, 32), c);
                int n = t * 16 + lrow;
#pragma unroll
                for (int rr = 0; rr < 8; ++rr) {
                    float hv = fmaxf(c[rr] + sBias2[n], 0.f);
                    stagw[(rr + 8 * g) * 80 + n] = (_Float16)hv;
                }
            }
        }
        // ---- layer 3: 64 -> 64, 8 WMMAs ----
        {
            v16h alo = loadA(stag, 80, lrow, g, 0);
            v16h ahi = loadA(stag, 80, lrow, g, 32);
#pragma unroll
            for (int t = 0; t < 4; ++t) {
                v8f c = {};
                c = wmma16(alo, loadB(&sB3[0][0], 80, t * 16 + lrow, g, 0), c);
                c = wmma16(ahi, loadB(&sB3[0][0], 80, t * 16 + lrow, g, 32), c);
                int n = t * 16 + lrow;
#pragma unroll
                for (int rr = 0; rr < 8; ++rr) {
                    float hv = fmaxf(c[rr] + sBias3[n], 0.f);
                    stagw[(rr + 8 * g) * 80 + n] = (_Float16)hv;
                }
            }
        }
        // ---- heads: 64 -> 18(pad32), 4 WMMAs; logits (f32) alias the staging ----
        {
            v16h alo = loadA(stag, 80, lrow, g, 0);
            v16h ahi = loadA(stag, 80, lrow, g, 32);
            float* lbase = (float*)&sStag[wave][0][0];   // row stride = 40 floats
#pragma unroll
            for (int t = 0; t < 2; ++t) {
                v8f c = {};
                c = wmma16(alo, loadB(&sBh[0][0], 80, t * 16 + lrow, g, 0), c);
                c = wmma16(ahi, loadB(&sBh[0][0], 80, t * 16 + lrow, g, 32), c);
                int n = t * 16 + lrow;
#pragma unroll
                for (int rr = 0; rr < 8; ++rr)
                    lbase[(rr + 8 * g) * 40 + n] = c[rr] + sBiasH[n];
            }
        }

        // ---- per-row gumbel-hard decisions ----
        if (lane < 16) {
            int r = waveRow + lane;
            const float* lp = (const float*)&sStag[wave][lane][0];
            int grp = (r >> 6) & 1;
#pragma unroll
            for (int j = 0; j < 6; ++j) atomicAdd(&sLyAcc[grp][j], lp[j]);
            float bestH = -1e30f, bestD = -1e30f;
            int iH = 0, iD = 0;
#pragma unroll
            for (int j = 0; j < 6; ++j) {
                float zH = (lp[6 + j]  + gumbelN(2u, (unsigned)(r * 6 + j))) * tauInv;
                float zD = (lp[12 + j] + gumbelN(3u, (unsigned)(r * 6 + j))) * tauInv;
                if (zH > bestH) { bestH = zH; iH = j; }
                if (zD > bestD) { bestD = zD; iD = j; }
            }
            int lr = wave * 16 + lane;
            sIdxH[lr] = iH; sIdxD[lr] = iD;
        }
        __syncthreads();

        if (tid < 2) {                                      // y head: mean over K then gumbel
            int bl = blBase + tid;
            float best = -1e30f; int iy = 0;
#pragma unroll
            for (int j = 0; j < 6; ++j) {
                float z = (sLyAcc[tid][j] * (1.0f / 64.0f) +
                           gumbelN(1u, (unsigned)(bl * 6 + j))) * tauInv;
                if (z > best) { best = z; iy = j; }
            }
            sIY[tid] = iy;
            float yb = 16.0f * sBitsTab[iy];
            sYbits[tid] = yb;
            out_yb[bl] = yb;
#pragma unroll
            for (int j = 0; j < 6; ++j) out_wy[bl * 6 + j] = (j == iy) ? 1.0f : 0.0f;
        }
        __syncthreads();

        // ---- per-row small outputs ----
        if (tid < 128) {
            int r = rowBase + tid, grp = tid >> 6;
            int iH = sIdxH[tid], iD = sIdxD[tid];
            float sD = sScale[12 + iD], qnD = sQNTab[iD], qpD = sQPTab[iD];
            float2 vv = ((const float2*)v)[r];
            float2 q;
            q.x = lsqf(vv.x, sD, qnD, qpD);
            q.y = lsqf(vv.y, sD, qnD, qpD);
            ((float2*)out_vq)[r] = q;
            float Hb = 16.0f * sBitsTab[iH];
            float Db = 2.0f  * sBitsTab[iD];
            out_Hb[r] = Hb;
            out_db[r] = Db;
            out_eb[r] = sYbits[grp] * (1.0f / 64.0f) + Hb + Db;
#pragma unroll
            for (int j = 0; j < 6; ++j) {
                out_wH[(size_t)r * 6 + j] = (j == iH) ? 1.0f : 0.0f;
                out_wd[(size_t)r * 6 + j] = (j == iD) ? 1.0f : 0.0f;
            }
        } else if (tid < 160) {                              // y_q: 16 values per (b,l)
            int t2 = tid - 128, grp = t2 >> 4, j = t2 & 15;
            int bl = blBase + grp, iy = sIY[grp];
            float s = sScale[iy], qn = sQNTab[iy], qp = sQPTab[iy];
            float x = y[(size_t)bl * 16 + j];
            out_yq[(size_t)bl * 16 + j] = lsqf(x, s, qn, qp);
        }

        // ---- coalesced H_q pass: float2 over (n,k), k contiguous ----
#pragma unroll
        for (int grp = 0; grp < 2; ++grp) {
            int bl = blBase + grp;
#pragma unroll
            for (int it = 0; it < 2; ++it) {
                int n = it * 4 + (tid >> 6), k = tid & 63;
                int iH = sIdxH[grp * 64 + k];
                float s = sScale[6 + iH], qn = sQNTab[iH], qp = sQPTab[iH];
                size_t fi = (size_t)bl * 512 + (size_t)n * 64 + k;
                float2 hv = ((const float2*)H)[fi];
                float2 hq;
                hq.x = lsqf(hv.x, s, qn, qp);
                hq.y = lsqf(hv.y, s, qn, qp);
                ((float2*)out_Hq)[fi] = hq;
            }
        }
        __syncthreads();   // protect shared reuse across chunks
    }
}

// ---------------- host entry ----------------
extern "C" void kernel_launch(void* const* d_in, const int* in_sizes, int n_in,
                              void* d_out, int out_size, void* d_ws, size_t ws_size,
                              hipStream_t stream) {
    const float* v   = (const float*)d_in[0];
    const float* H   = (const float*)d_in[1];
    const float* y   = (const float*)d_in[2];
    const float* snr = (const float*)d_in[3];
    const float* w1  = (const float*)d_in[4];
    const float* b1  = (const float*)d_in[5];
    const float* w2  = (const float*)d_in[6];
    const float* b2  = (const float*)d_in[7];
    const float* w3  = (const float*)d_in[8];
    const float* b3  = (const float*)d_in[9];
    const float* wy  = (const float*)d_in[10];
    const float* by  = (const float*)d_in[11];
    const float* wh  = (const float*)d_in[12];
    const float* bh  = (const float*)d_in[13];
    const float* wd  = (const float*)d_in[14];
    const float* bd  = (const float*)d_in[15];
    const float* s_y = (const float*)d_in[16];
    const float* s_H = (const float*)d_in[17];
    const float* s_d = (const float*)d_in[18];
    const int*   tau = (const int*)d_in[19];

    // workspace layout (floats): h_power[1M] | total_power[16K] | y_power[16K] | avg[4K]
    float* ws     = (float*)d_ws;
    float* ws_hp  = ws;
    float* ws_tp  = ws + 1048576;
    float* ws_yp  = ws + 1048576 + 16384;
    float* ws_avg = ws + 1048576 + 32768;

    taq_prep_kernel<<<B_ * L_, K_, 0, stream>>>(H, y, ws_hp, ws_tp, ws_yp);
    taq_avg_kernel<<<(B_ * K_) / 256, 256, 0, stream>>>(ws_hp, ws_avg);
    taq_main_kernel<<<NBLOCKS, 256, 0, stream>>>(
        v, H, y, snr, w1, b1, w2, b2, w3, b3, wy, by, wh, bh, wd, bd,
        s_y, s_H, s_d, tau, ws_hp, ws_tp, ws_yp, ws_avg, (float*)d_out);
}
